// DAGSparseSelfAttention_49108656062888
// MI455X (gfx1250) — matrix operand order repaired and checked by reading, not compiled, not run on hardware
//
#include <hip/hip_runtime.h>
#include <hip/hip_bf16.h>
#include <math.h>

typedef __attribute__((ext_vector_type(16))) _Float16 v16h;
typedef __attribute__((ext_vector_type(8)))  _Float16 v8h;
typedef __attribute__((ext_vector_type(4)))  _Float16 v4h;
typedef __attribute__((ext_vector_type(8)))  float    v8f;

__device__ __forceinline__ v8f wmma_f16(v16h a, v16h b, v8f c) {
  // D = A(16x32 f16) * B(32x16 f16) + C(16x16 f32)
  return __builtin_amdgcn_wmma_f32_16x16x32_f16(
      /*neg_a=*/false, a, /*neg_b=*/false, b,
      /*c_mod=*/(short)0, c, /*reuse_a=*/false, /*reuse_b=*/false);
}

__device__ __forceinline__ float gelu_exact(float x) {
  return 0.5f * x * (1.0f + erff(x * 0.70710678118654752f));
}

// 32-bit LDS byte offset of a __shared__ object (generic -> AS(3) cast).
__device__ __forceinline__ unsigned lds_off(const void* p) {
  return (unsigned)(unsigned long long)
      (__attribute__((address_space(3))) const void*)p;
}

// Issue one per-lane async copy of 16 bytes global -> LDS (ASYNCcnt).
__device__ __forceinline__ void async_copy_b128(unsigned lds_addr,
                                                const void* gaddr) {
  asm volatile("global_load_async_to_lds_b128 %0, %1, off"
               :: "v"(lds_addr), "v"(gaddr) : "memory");
}
__device__ __forceinline__ void wait_asynccnt0() {
  asm volatile("s_wait_asynccnt 0x0" ::: "memory");
}

// Build the 16-element A fragment from two contiguous 8-half runs
// (K = kb..kb+7 and kb+8..kb+15 for this lane, kb = half*16).
__device__ __forceinline__ v16h load_a_frag(const _Float16* ar) {
  v8h a0 = *(const v8h*)ar;
  v8h a1 = *(const v8h*)(ar + 8);
  v16h a;
#pragma unroll
  for (int e = 0; e < 8; ++e) { a[e] = a0[e]; a[e + 8] = a1[e]; }
  return a;
}

// ---------------------------------------------------------------------------
// f32 -> f16 conversion pre-pass (4 elems / thread)
// ---------------------------------------------------------------------------
__global__ __launch_bounds__(256) void cvt_f16_kernel(
    const float* __restrict__ src, _Float16* __restrict__ dst, int n) {
  const int i = (blockIdx.x * 256 + threadIdx.x) * 4;
  if (i >= n) return;
  const float4 f = *(const float4*)(src + i);
  v4h h;
  h[0] = (_Float16)f.x; h[1] = (_Float16)f.y;
  h[2] = (_Float16)f.z; h[3] = (_Float16)f.w;
  *(v4h*)(dst + i) = h;
}

// ---------------------------------------------------------------------------
// Tiled GEMM, f16 in / f32 accumulate: out = concat(A0,A1)[M,K] @ W[N,K]^T + b
// One wave computes a 16x64 tile (4 accumulators share one A fragment).
// ---------------------------------------------------------------------------
template <bool HALF_OUT>
__global__ __launch_bounds__(256) void gemm16x64(
    const _Float16* __restrict__ A0, const _Float16* __restrict__ A1, int Ksplit,
    const _Float16* __restrict__ W, const float* __restrict__ bias,
    float* __restrict__ outF, _Float16* __restrict__ outH,
    int M, int N, int K) {
  const int wid  = threadIdx.x >> 5;
  const int lane = threadIdx.x & 31;
  const int half = lane >> 4;
  const int lan  = lane & 15;

  const int ntiles = N >> 6;                    // 64-wide tiles
  const int tile = blockIdx.x * 8 + wid;
  if (tile >= (M >> 4) * ntiles) return;
  const int m0 = (tile / ntiles) << 4;
  const int n0 = (tile % ntiles) << 6;

  v8f acc0 = {}, acc1 = {}, acc2 = {}, acc3 = {};
  for (int k0 = 0; k0 < K; k0 += 32) {
    const _Float16* Ap; int kk, lda;
    if (k0 < Ksplit) { Ap = A0; kk = k0;          lda = Ksplit;     }
    else             { Ap = A1; kk = k0 - Ksplit; lda = K - Ksplit; }

    const v16h a =
        load_a_frag(Ap + (size_t)(m0 + lan) * lda + kk + half * 16);

    // B fragments: B[k][n] = W[n][k]; lane holds col, 16 contiguous K halves
    const _Float16* br = W + (size_t)(n0 + lan) * K + k0 + half * 16;
    const size_t wstride = (size_t)16 * K;      // next 16-column group
    if (k0 + 32 < K) __builtin_prefetch(br + 32, 0, 3);   // global_prefetch_b8
    acc0 = wmma_f16(a, *(const v16h*)(br),               acc0);
    acc1 = wmma_f16(a, *(const v16h*)(br + wstride),     acc1);
    acc2 = wmma_f16(a, *(const v16h*)(br + 2 * wstride), acc2);
    acc3 = wmma_f16(a, *(const v16h*)(br + 3 * wstride), acc3);
  }

#pragma unroll
  for (int j = 0; j < 4; ++j) {
    const v8f acc = (j == 0) ? acc0 : (j == 1) ? acc1 : (j == 2) ? acc2 : acc3;
    const int n = n0 + j * 16 + lan;
    const float bs = bias ? bias[n] : 0.0f;
#pragma unroll
    for (int r = 0; r < 8; ++r) {
      const int m = m0 + r + half * 8;
      const float val = acc[r] + bs;
      if (HALF_OUT) outH[(size_t)m * N + n] = (_Float16)val;
      else          outF[(size_t)m * N + n] = val;
    }
  }
}

// ---------------------------------------------------------------------------
// Flash-style masked attention, q/k/v f16. One block per (b, h, 16 query
// rows); 128-key chunks. v chunk staged transposed in LDS (vector ds loads),
// mask chunk staged via async global->LDS b128 copies (ASYNCcnt).
// ---------------------------------------------------------------------------
#define L_SEQ 1024
#define HD 32
#define CHUNK 128

__global__ __launch_bounds__(256) void attn_kernel(
    const _Float16* __restrict__ q, const _Float16* __restrict__ k,
    const _Float16* __restrict__ v, const int* __restrict__ mask,
    float* __restrict__ y) {
  const int bid = blockIdx.x;
  const int lt = bid & 63;
  const int h  = (bid >> 6) & 7;
  const int b  = bid >> 9;
  const int l0 = lt << 4;

  const int tid  = threadIdx.x;
  const int wid  = tid >> 5;
  const int lane = tid & 31;
  const int half = lane >> 4;
  const int lan  = lane & 15;

  __shared__ float    s_sc[16][CHUNK];     // masked scores
  __shared__ _Float16 s_p[16][CHUNK];      // probabilities (A of p@v)
  __shared__ _Float16 s_v[HD][CHUNK];      // v chunk, transposed [d][token]
  __shared__ int      s_mk[16][CHUNK];     // mask chunk (async-staged)
  __shared__ float    s_yacc[16][HD];
  __shared__ float    s_pm[16][16], s_ps[16][16];
  __shared__ float    s_m[16], s_l[16], s_alpha[16];

  if (tid < 16) { s_m[tid] = -3.0e38f; s_l[tid] = 0.0f; }
  for (int i = tid; i < 16 * HD; i += 256) ((float*)s_yacc)[i] = 0.0f;

  // q A-fragment, loaded once
  const v16h afrag =
      load_a_frag(q + (size_t)(b * L_SEQ + l0 + lan) * 256 + h * HD + half * 16);

  __syncthreads();

  for (int c0 = 0; c0 < L_SEQ; c0 += CHUNK) {
    // ---- stage mask chunk via async global->LDS (2 x b128 per thread) ----
    {
      const int row  = tid >> 4;               // 0..15 query rows
      const int col0 = (tid & 15) * 8;         // 8 ints = 32B per thread
      const int* gsrc =
          mask + ((size_t)b * L_SEQ + l0 + row) * L_SEQ + c0 + col0;
      const unsigned l = lds_off(&s_mk[row][col0]);
      async_copy_b128(l,       gsrc);
      async_copy_b128(l + 16u, gsrc + 4);
    }
    // ---- stage v chunk transposed: s_v[d][tok] (coalesced 32B loads) ----
    {
      const int tloc = tid >> 1;               // 0..127
      const int dpart = (tid & 1) * 16;
      const _Float16* vv =
          v + (size_t)(b * L_SEQ + c0 + tloc) * 256 + h * HD + dpart;
      const v8h v0 = *(const v8h*)vv;
      const v8h v1 = *(const v8h*)(vv + 8);
#pragma unroll
      for (int e = 0; e < 8; ++e) {
        s_v[dpart + e][tloc]     = v0[e];
        s_v[dpart + 8 + e][tloc] = v1[e];
      }
    }
    wait_asynccnt0();
    __syncthreads();                            // mask + v staged

    // ---- 1) scores: wave w -> 16x16 tile at key cols c0 + w*16 ----
    {
      const int ccol = c0 + wid * 16;
      const v16h bfrag = *(const v16h*)(
          k + (size_t)(b * L_SEQ + ccol + lan) * 256 + h * HD + half * 16);
      v8f sc = {};
      sc = wmma_f16(afrag, bfrag, sc);
#pragma unroll
      for (int r = 0; r < 8; ++r) {
        const int mrow = r + half * 8;
        const int lrow = l0 + mrow;
        const int mcol = ccol + lan;
        const int mk = s_mk[mrow][wid * 16 + lan];
        const bool dead = (mk == 0) || (mcol == lrow);
        s_sc[mrow][wid * 16 + lan] = dead ? -3.0e38f : sc[r];
      }
    }
    __syncthreads();

    // ---- 2) row max (online) ----
    {
      const int row = tid >> 4, grp = tid & 15;
      float pm = -3.0e38f;
#pragma unroll
      for (int j = 0; j < 8; ++j) pm = fmaxf(pm, s_sc[row][grp * 8 + j]);
      s_pm[row][grp] = pm;
    }
    __syncthreads();
    if (tid < 16) {
      float mo = s_m[tid], mn = mo;
#pragma unroll
      for (int g = 0; g < 16; ++g) mn = fmaxf(mn, s_pm[tid][g]);
      s_alpha[tid] = __expf(mo - mn);
      s_m[tid] = mn;
    }
    __syncthreads();

    // ---- 3) rescale accumulator, p = exp(s - m) (masked -> 0) ----
    {
#pragma unroll
      for (int i = 0; i < 2; ++i) {
        const int idx = tid + i * 256;
        ((float*)s_yacc)[idx] *= s_alpha[idx >> 5];
      }
      const int row = tid >> 4, grp = tid & 15;
      const float mn = s_m[row];
      float ps = 0.0f;
#pragma unroll
      for (int j = 0; j < 8; ++j) {
        const float x = s_sc[row][grp * 8 + j];
        const float p = (x <= -2.9e38f) ? 0.0f : __expf(x - mn);
        ps += p;
        s_p[row][grp * 8 + j] = (_Float16)p;
      }
      s_ps[row][grp] = ps;
    }
    __syncthreads();
    if (tid < 16) {
      float ls = 0.0f;
#pragma unroll
      for (int g = 0; g < 16; ++g) ls += s_ps[tid][g];
      s_l[tid] = s_l[tid] * s_alpha[tid] + ls;
    }

    // ---- 4) y += p(16x128) @ v(128x32): wave w -> n-tile w&1, K-quarter w>>1
    {
      const int n0 = (wid & 1) * 16;
      const int kq = wid >> 1;
      const v16h pa = load_a_frag(&s_p[lan][kq * 32 + half * 16]);
      const v16h vb =
          *(const v16h*)&s_v[n0 + lan][kq * 32 + half * 16];
      v8f pc = {};
      pc = wmma_f16(pa, vb, pc);
#pragma unroll
      for (int r = 0; r < 8; ++r)
        atomicAdd(&s_yacc[r + half * 8][n0 + lan], pc[r]);
    }
    __syncthreads();
  }

  for (int i = tid; i < 16 * HD; i += 256) {
    const int row = i >> 5, d = i & 31;
    const float val = s_yacc[row][d] / fmaxf(s_l[row], 1e-20f);
    y[(size_t)(b * L_SEQ + l0 + row) * 256 + h * HD + d] = val;
  }
}

// ---------------------------------------------------------------------------
// out[row] = layernorm(gelu(X[row])) * g + b, one wave per row (D = 256)
// ---------------------------------------------------------------------------
__global__ __launch_bounds__(256) void rowln_gelu(
    const float* __restrict__ X, const float* __restrict__ g,
    const float* __restrict__ b, float* __restrict__ out, int D) {
  const int row  = blockIdx.x * 8 + (threadIdx.x >> 5);
  const int lane = threadIdx.x & 31;
  const float* xr = X + (size_t)row * D;
  const int per = D >> 5;
  float vals[16];
  float s = 0.0f;
  for (int i = 0; i < per; ++i) {
    float x = gelu_exact(xr[lane + i * 32]);
    vals[i] = x; s += x;
  }
  for (int off = 16; off; off >>= 1) s += __shfl_xor(s, off, 32);
  const float mean = s / (float)D;
  float vs = 0.0f;
  for (int i = 0; i < per; ++i) { float d = vals[i] - mean; vs += d * d; }
  for (int off = 16; off; off >>= 1) vs += __shfl_xor(vs, off, 32);
  const float rstd = rsqrtf(vs / (float)D + 1e-5f);
  for (int i = 0; i < per; ++i) {
    const int c = lane + i * 32;
    out[(size_t)row * D + c] = (vals[i] - mean) * rstd * g[c] + b[c];
  }
}

// ---------------------------------------------------------------------------
// zn[row] = layernorm(concat(Y[row], O2[row])) * g + b  -> f16 (512 wide)
// ---------------------------------------------------------------------------
__global__ __launch_bounds__(256) void concat_ln_h(
    const float* __restrict__ Y, const float* __restrict__ O2,
    const float* __restrict__ g, const float* __restrict__ b,
    _Float16* __restrict__ out) {
  const int row  = blockIdx.x * 8 + (threadIdx.x >> 5);
  const int lane = threadIdx.x & 31;
  const float* yr = Y  + (size_t)row * 256;
  const float* o2 = O2 + (size_t)row * 256;
  float vals[16];
  float s = 0.0f;
#pragma unroll
  for (int i = 0; i < 16; ++i) {
    const int c = lane + i * 32;
    const float x = (c < 256) ? yr[c] : o2[c - 256];
    vals[i] = x; s += x;
  }
  for (int off = 16; off; off >>= 1) s += __shfl_xor(s, off, 32);
  const float mean = s * (1.0f / 512.0f);
  float vs = 0.0f;
#pragma unroll
  for (int i = 0; i < 16; ++i) { float d = vals[i] - mean; vs += d * d; }
  for (int off = 16; off; off >>= 1) vs += __shfl_xor(vs, off, 32);
  const float rstd = rsqrtf(vs * (1.0f / 512.0f) + 1e-5f);
#pragma unroll
  for (int i = 0; i < 16; ++i) {
    const int c = lane + i * 32;
    out[(size_t)row * 512 + c] =
        (_Float16)((vals[i] - mean) * rstd * g[c] + b[c]);
  }
}

// ---------------------------------------------------------------------------
extern "C" void kernel_launch(void* const* d_in, const int* in_sizes, int n_in,
                              void* d_out, int out_size, void* d_ws, size_t ws_size,
                              hipStream_t stream) {
  const float* obs   = (const float*)d_in[0];
  const float* act   = (const float*)d_in[1];
  const int*   msk   = (const int*)  d_in[2];
  const float* Wq    = (const float*)d_in[3];
  const float* bq    = (const float*)d_in[4];
  const float* Wk    = (const float*)d_in[5];
  const float* bk    = (const float*)d_in[6];
  const float* Wv    = (const float*)d_in[7];
  const float* bv    = (const float*)d_in[8];
  const float* Wobs  = (const float*)d_in[9];
  const float* bobs  = (const float*)d_in[10];
  const float* g_obs = (const float*)d_in[11];
  const float* b_obs = (const float*)d_in[12];
  const float* g1    = (const float*)d_in[13];
  const float* b1    = (const float*)d_in[14];
  const float* Wp    = (const float*)d_in[15];
  const float* bp    = (const float*)d_in[16];
  const float* g2    = (const float*)d_in[17];
  const float* b2    = (const float*)d_in[18];
  float* out = (float*)d_out;

  // ---- carve workspace (256B-aligned chunks) ----
  char* base = (char*)d_ws;
  auto alloc = [&](size_t bytes) -> char* {
    char* p = base;
    base += (bytes + 255) & ~(size_t)255;
    return p;
  };
  const size_t Rows = 8192;                  // B*L
  _Float16* obs_h  = (_Float16*)alloc(Rows * 256 * 2);
  _Float16* act_h  = (_Float16*)alloc(Rows * 256 * 2);
  _Float16* Wq_h   = (_Float16*)alloc(256 * 256 * 2);
  _Float16* Wk_h   = (_Float16*)alloc(256 * 512 * 2);
  _Float16* Wv_h   = (_Float16*)alloc(256 * 512 * 2);
  _Float16* Wobs_h = (_Float16*)alloc(256 * 256 * 2);
  _Float16* Wp_h   = (_Float16*)alloc(256 * 512 * 2);
  _Float16* qh     = (_Float16*)alloc(Rows * 256 * 2);
  _Float16* kh     = (_Float16*)alloc(Rows * 256 * 2);
  _Float16* vh     = (_Float16*)alloc(Rows * 256 * 2);
  _Float16* znh    = (_Float16*)alloc(Rows * 512 * 2);
  float* pre    = (float*)alloc(Rows * 256 * 4);
  float* obs2   = (float*)alloc(Rows * 256 * 4);
  float* yb     = (float*)alloc(Rows * 256 * 4);
  float* outpre = (float*)alloc(Rows * 256 * 4);

  const dim3 blk(256);
  auto cvt = [&](const float* s, _Float16* d, int n) {
    cvt_f16_kernel<<<n / 1024, blk, 0, stream>>>(s, d, n);
  };
  cvt(obs,  obs_h,  Rows * 256);
  cvt(act,  act_h,  Rows * 256);
  cvt(Wq,   Wq_h,   256 * 256);
  cvt(Wk,   Wk_h,   256 * 512);
  cvt(Wv,   Wv_h,   256 * 512);
  cvt(Wobs, Wobs_h, 256 * 256);
  cvt(Wp,   Wp_h,   256 * 512);

  // GEMMs: (8192/16)*(256/64) = 2048 wave-tiles / 8 -> 256 blocks
  gemm16x64<true ><<<256, blk, 0, stream>>>(obs_h, nullptr, 256, Wq_h,   bq,   nullptr, qh,     8192, 256, 256);
  gemm16x64<true ><<<256, blk, 0, stream>>>(obs_h, act_h,   256, Wk_h,   bk,   nullptr, kh,     8192, 256, 512);
  gemm16x64<true ><<<256, blk, 0, stream>>>(obs_h, act_h,   256, Wv_h,   bv,   nullptr, vh,     8192, 256, 512);
  gemm16x64<false><<<256, blk, 0, stream>>>(obs_h, nullptr, 256, Wobs_h, bobs, pre,     nullptr, 8192, 256, 256);
  rowln_gelu<<<1024, blk, 0, stream>>>(pre, g_obs, b_obs, obs2, 256);

  attn_kernel<<<4096, blk, 0, stream>>>(qh, kh, vh, msk, yb);

  concat_ln_h<<<1024, blk, 0, stream>>>(yb, obs2, g1, b1, znh);
  gemm16x64<false><<<256, blk, 0, stream>>>(znh, nullptr, 512, Wp_h, bp, outpre, nullptr, 8192, 256, 512);
  rowln_gelu<<<1024, blk, 0, stream>>>(outpre, g2, b2, out, 256);
}
